// CausalSelfAttention_82145544503868
// MI455X (gfx1250) — compile-verified, matrix operand order checked by reading, not run on hardware
//
#include <hip/hip_runtime.h>
#include <hip/hip_bf16.h>

// ---------------------------------------------------------------------------
// CDNA5 (gfx1250) causal self-attention, bf16 WMMA with fp32 accumulation.
//   phase 1: qkv = x @ w_qkv   (fp32 -> bf16 staged in LDS, WMMA; V stored ^T)
//   phase 2: flash attention   (K tiles via async global->LDS DMA, double buf)
//   phase 3: out = O @ w_out   (bf16 A via async LDS copy, WMMA, fp32 out)
// ---------------------------------------------------------------------------

typedef __attribute__((ext_vector_type(16))) __bf16 v16bf;
typedef __attribute__((ext_vector_type(8)))  float  v8f;

union Frag {
    v16bf        v;
    unsigned int u[8];
};

__device__ __forceinline__ unsigned short f2bf(float f) {
    unsigned int u = __float_as_uint(f);
    unsigned int r = u + 0x7fffu + ((u >> 16) & 1u);   // round-to-nearest-even
    return (unsigned short)(r >> 16);
}

// K-index pattern of the 16-bit A/B WMMA fragment (CDNA5 ISA):
// lanes 0-15 hold K = {0..7} then {16..23}; lanes 16-31 hold K+8.
__device__ __forceinline__ int kpat(int lane, int j) {
    int half = (lane >> 4) * 8;
    return (j < 4) ? (half + 2 * j) : (16 + half + 2 * (j - 4));
}

__device__ __forceinline__ v8f wmma_bf16(const Frag& a, const Frag& b, v8f c) {
    return __builtin_amdgcn_wmma_f32_16x16x32_bf16(
        false, a.v, false, b.v, (short)0, c, false, false);
}

// ---- 16-lane xor-butterfly reductions on the VALU (v_permlane16_b32) ------
#if __has_builtin(__builtin_amdgcn_permlane16)
__device__ __forceinline__ float xor16(float x, unsigned s0, unsigned s1) {
    return __uint_as_float(__builtin_amdgcn_permlane16(
        __float_as_uint(x), __float_as_uint(x), s0, s1, false, false));
}
__device__ __forceinline__ float rowmax16(float x) {
    x = fmaxf(x, xor16(x, 0x67452301u, 0xEFCDAB89u));   // xor 1
    x = fmaxf(x, xor16(x, 0x54761032u, 0xDCFE98BAu));   // xor 2
    x = fmaxf(x, xor16(x, 0x32107654u, 0xBA98FEDCu));   // xor 4
    x = fmaxf(x, xor16(x, 0xFEDCBA98u, 0x76543210u));   // xor 8
    return x;
}
__device__ __forceinline__ float rowsum16(float x) {
    x += xor16(x, 0x67452301u, 0xEFCDAB89u);
    x += xor16(x, 0x54761032u, 0xDCFE98BAu);
    x += xor16(x, 0x32107654u, 0xBA98FEDCu);
    x += xor16(x, 0xFEDCBA98u, 0x76543210u);
    return x;
}
#else
__device__ __forceinline__ float rowmax16(float x) {
    x = fmaxf(x, __shfl_xor(x, 1, 32));
    x = fmaxf(x, __shfl_xor(x, 2, 32));
    x = fmaxf(x, __shfl_xor(x, 4, 32));
    x = fmaxf(x, __shfl_xor(x, 8, 32));
    return x;
}
__device__ __forceinline__ float rowsum16(float x) {
    x += __shfl_xor(x, 1, 32);
    x += __shfl_xor(x, 2, 32);
    x += __shfl_xor(x, 4, 32);
    x += __shfl_xor(x, 8, 32);
    return x;
}
#endif

// ---- async global->LDS 16B copy (ASYNCcnt-tracked DMA, gfx1250) -----------
__device__ __forceinline__ void async_b128_to_lds(unsigned lds_addr,
                                                  unsigned long long gaddr) {
    asm volatile("global_load_async_to_lds_b128 %0, %1, off"
                 :: "v"(lds_addr), "v"(gaddr) : "memory");
}

// ---------------------------------------------------------------------------
// Generic 128x128-tile GEMM:  C[M,N] = A[M,K] * B[K,N]
//   MODE 0: A fp32; epilogue scatters bf16 Q/K [B,H,T,64] and V^T [B,H,64,T]
//           (Q scaled by 1/sqrt(64) = 0.125)
//   MODE 1: A bf16 (async-copied to LDS); epilogue writes fp32 to outf
// 8 waves per block (wave32), wave grid 4(M) x 2(N), wave tile 32x64.
// ---------------------------------------------------------------------------
template <int MODE>
__global__ __launch_bounds__(256) void gemm_bf16_wmma(
    const void* __restrict__ Aptr, const float* __restrict__ Bptr,
    int M, int N, int K, int lda, int ldb,
    unsigned short* __restrict__ qbuf, unsigned short* __restrict__ kbuf,
    unsigned short* __restrict__ vbufT, float* __restrict__ outf) {

    __shared__ alignas(16) unsigned short As[128 * 32];  // [m][k] bf16
    __shared__ alignas(16) unsigned short Bs[128 * 32];  // [n][k] bf16

    const int tid  = threadIdx.x;
    const int lane = tid & 31;
    const int wid  = tid >> 5;
    const int wm   = wid >> 1;        // 0..3
    const int wn   = wid & 1;         // 0..1
    const int row0 = blockIdx.y * 128;
    const int col0 = blockIdx.x * 128;

    v8f acc[2][4] = {};

    for (int k0 = 0; k0 < K; k0 += 32) {
        __syncthreads();   // protect LDS from previous iteration's readers

        // ---- stage A tile: 128 rows x 32 k ----
        if (MODE == 0) {
            #pragma unroll
            for (int i = 0; i < 4; ++i) {
                int idx = tid + i * 256;          // 0..1023
                int r   = idx >> 3;               // row 0..127
                int c4  = (idx & 7) * 4;          // k chunk of 4
                const float4 a = *(const float4*)((const float*)Aptr +
                                    (size_t)(row0 + r) * lda + k0 + c4);
                uint2 p;
                p.x = (unsigned)f2bf(a.x) | ((unsigned)f2bf(a.y) << 16);
                p.y = (unsigned)f2bf(a.z) | ((unsigned)f2bf(a.w) << 16);
                *(uint2*)&As[r * 32 + c4] = p;
            }
        } else {
            // bf16 A: raw 16B chunks via async DMA straight into LDS
            const unsigned long long abase =
                (unsigned long long)(size_t)((const unsigned short*)Aptr) +
                ((size_t)row0 * lda + k0) * 2ull;
            unsigned ldsbase = (unsigned)(size_t)&As[0];
            #pragma unroll
            for (int i = 0; i < 2; ++i) {
                int idx = tid + i * 256;          // chunk 0..511 (16B each)
                int r   = idx >> 2;               // row 0..127
                int c   = (idx & 3) * 16;         // byte offset in 64B row
                async_b128_to_lds(ldsbase + (unsigned)(r * 64 + c),
                                  abase + (size_t)r * lda * 2ull + c);
            }
            asm volatile("s_wait_asynccnt 0" ::: "memory");
        }
        // ---- stage B tile: 32 k-rows x 128 n, transposed into [n][k] ----
        #pragma unroll
        for (int i = 0; i < 4; ++i) {
            int idx = tid + i * 256;          // 0..1023
            int kr  = idx >> 5;               // k row 0..31
            int nc  = (idx & 31) * 4;         // n chunk of 4
            const float4 b = *(const float4*)(Bptr + (size_t)(k0 + kr) * ldb +
                                              col0 + nc);
            Bs[(nc + 0) * 32 + kr] = f2bf(b.x);
            Bs[(nc + 1) * 32 + kr] = f2bf(b.y);
            Bs[(nc + 2) * 32 + kr] = f2bf(b.z);
            Bs[(nc + 3) * 32 + kr] = f2bf(b.w);
        }

        // hint the next K tile into cache while we compute
        if (k0 + 32 < K) {
            if (MODE == 0)
                __builtin_prefetch((const float*)Aptr +
                                   (size_t)(row0 + (tid >> 1)) * lda + k0 + 32, 0, 0);
            __builtin_prefetch(Bptr + (size_t)(k0 + 32 + (tid >> 4)) * ldb + col0, 0, 0);
        }

        __syncthreads();

        // ---- fragments + 8 WMMAs ----
        Frag afr[2];
        #pragma unroll
        for (int mt = 0; mt < 2; ++mt) {
            int r = wm * 32 + mt * 16 + (lane & 15);
            #pragma unroll
            for (int j = 0; j < 8; ++j)
                afr[mt].u[j] = *(const unsigned int*)&As[r * 32 + kpat(lane, j)];
        }
        #pragma unroll
        for (int nt = 0; nt < 4; ++nt) {
            Frag bfr;
            int n = wn * 64 + nt * 16 + (lane & 15);
            #pragma unroll
            for (int j = 0; j < 8; ++j)
                bfr.u[j] = *(const unsigned int*)&Bs[n * 32 + kpat(lane, j)];
            #pragma unroll
            for (int mt = 0; mt < 2; ++mt)
                acc[mt][nt] = wmma_bf16(afr[mt], bfr, acc[mt][nt]);
        }
    }

    // ---- epilogue ----
    #pragma unroll
    for (int mt = 0; mt < 2; ++mt) {
        #pragma unroll
        for (int nt = 0; nt < 4; ++nt) {
            #pragma unroll
            for (int r = 0; r < 8; ++r) {
                int grow = row0 + wm * 32 + mt * 16 + r + 8 * (lane >> 4);
                int gcol = col0 + wn * 64 + nt * 16 + (lane & 15);
                float val = acc[mt][nt][r];
                if (MODE == 0) {
                    int b  = grow >> 11;          // / 2048
                    int t  = grow & 2047;
                    int s  = gcol >> 10;          // 0:q 1:k 2:v
                    int cc = gcol & 1023;
                    int h  = cc >> 6;
                    int dh = cc & 63;
                    int bh = b * 16 + h;
                    if (s == 0)
                        qbuf[((size_t)bh * 2048 + t) * 64 + dh] =
                            f2bf(val * 0.125f);                    // 1/sqrt(64)
                    else if (s == 1)
                        kbuf[((size_t)bh * 2048 + t) * 64 + dh] = f2bf(val);
                    else  // V transposed: [bh][dh][t] for contiguous-K PV loads
                        vbufT[((size_t)bh * 64 + dh) * 2048 + t] = f2bf(val);
                } else {
                    outf[(size_t)grow * N + gcol] = val;
                }
            }
        }
    }
}

// ---------------------------------------------------------------------------
// Flash attention: one wave per 16-query tile of one (b,h).
// Q/K bf16 [B*H,2048,64]; V^T bf16 [B*H,64,2048]; O bf16 [B,T,C].
// K tiles are double-buffered in LDS via async global->LDS DMA.
// ---------------------------------------------------------------------------
__global__ __launch_bounds__(256) void attn_flash_wmma(
    const unsigned short* __restrict__ Q, const unsigned short* __restrict__ Kb,
    const unsigned short* __restrict__ Vt, unsigned short* __restrict__ O) {

    __shared__ alignas(16) unsigned short Klds[8][2][32 * 64]; // per-wave K dbuf
    __shared__ alignas(16) unsigned short Plds[8][16 * 32];    // per-wave P tile

    const int tid  = threadIdx.x;
    const int lane = tid & 31;
    const int wid  = tid >> 5;
    const int gw   = blockIdx.x * 8 + wid;        // global wave id
    const int qt   = gw & 127;                    // q tile within (b,h)
    const int bh   = gw >> 7;                     // 0..63
    const int q0   = qt << 4;

    const unsigned short* Qb  = Q  + (size_t)bh * 2048 * 64;
    const unsigned short* Kbh = Kb + (size_t)bh * 2048 * 64;
    const unsigned short* Vbh = Vt + (size_t)bh * 64 * 2048;

    // async DMA of one 4KB K tile (32 keys x 64 dh, contiguous) into LDS
    auto issueK = [&](int kt, int buf) {
        const unsigned long long gb =
            (unsigned long long)(size_t)(Kbh + (size_t)kt * 64);
        unsigned ldsbase = (unsigned)(size_t)&Klds[wid][buf][0];
        #pragma unroll
        for (int i = 0; i < 8; ++i) {
            unsigned off = (unsigned)((i * 32 + lane) * 16);
            async_b128_to_lds(ldsbase + off, gb + off);
        }
    };

    // Q fragments for head-dim chunks kk=0..31 and 32..63 (held in registers)
    Frag qf[2];
    {
        const unsigned short* qr = Qb + (size_t)(q0 + (lane & 15)) * 64;
        #pragma unroll
        for (int kk = 0; kk < 2; ++kk)
            #pragma unroll
            for (int j = 0; j < 8; ++j)
                qf[kk].u[j] = *(const unsigned int*)&qr[kk * 32 + kpat(lane, j)];
    }

    v8f  o[4] = {};
    float mrow[8], lrow[8];
    #pragma unroll
    for (int r = 0; r < 8; ++r) { mrow[r] = -INFINITY; lrow[r] = 0.0f; }

    const int kend = q0 + 15;                     // causal bound
    issueK(0, 0);                                 // prefetch first K tile

    for (int kt0 = 0; kt0 <= kend; kt0 += 32) {
        const int  cur  = (kt0 >> 5) & 1;
        const bool more = (kt0 + 32) <= kend;
        if (more) {
            issueK(kt0 + 32, cur ^ 1);            // DMA next tile while computing
            asm volatile("s_wait_asynccnt 8" ::: "memory");  // current tile done
        } else {
            asm volatile("s_wait_asynccnt 0" ::: "memory");
        }

        // ---- S = Q @ K^T  (two 16x16 tiles over 32 keys; K from LDS) ----
        v8f s0 = {}, s1 = {};
        {
            const unsigned short* kr0 = &Klds[wid][cur][(lane & 15) * 64];
            const unsigned short* kr1 = kr0 + 16 * 64;
            Frag b0, b1;
            #pragma unroll
            for (int j = 0; j < 8; ++j) {
                int kp = kpat(lane, j);
                b0.u[j] = *(const unsigned int*)&kr0[kp];
                b1.u[j] = *(const unsigned int*)&kr0[32 + kp];
            }
            s0 = wmma_bf16(qf[0], b0, s0);
            s0 = wmma_bf16(qf[1], b1, s0);
            #pragma unroll
            for (int j = 0; j < 8; ++j) {
                int kp = kpat(lane, j);
                b0.u[j] = *(const unsigned int*)&kr1[kp];
                b1.u[j] = *(const unsigned int*)&kr1[32 + kp];
            }
            s1 = wmma_bf16(qf[0], b0, s1);
            s1 = wmma_bf16(qf[1], b1, s1);
        }

        // ---- causal mask + online softmax (rows live in 16-lane halves) ----
        const int nk0   = kt0 + (lane & 15);
        const int nk1   = nk0 + 16;
        const int rbase = q0 + 8 * (lane >> 4);
        #pragma unroll
        for (int r = 0; r < 8; ++r) {
            int   mr = rbase + r;
            float v0 = (nk0 <= mr) ? s0[r] : -1.0e30f;
            float v1 = (nk1 <= mr) ? s1[r] : -1.0e30f;
            float rmx  = rowmax16(fmaxf(v0, v1));
            float mnew = fmaxf(mrow[r], rmx);
            float al   = __expf(mrow[r] - mnew);
            float p0   = __expf(v0 - mnew);
            float p1   = __expf(v1 - mnew);
            float rs   = rowsum16(p0 + p1);
            lrow[r] = lrow[r] * al + rs;
            mrow[r] = mnew;
            #pragma unroll
            for (int nt = 0; nt < 4; ++nt) o[nt][r] *= al;

            int m = r + 8 * (lane >> 4);
            Plds[wid][m * 32 + (lane & 15)]      = f2bf(p0);
            Plds[wid][m * 32 + 16 + (lane & 15)] = f2bf(p1);
        }
        // intra-wave cross-lane LDS handoff: drain DS ops
        asm volatile("s_wait_dscnt 0" ::: "memory");

        // ---- rebuild P as an A fragment, then O += P @ V ----
        Frag pf;
        #pragma unroll
        for (int j = 0; j < 8; ++j)
            pf.u[j] = *(const unsigned int*)&Plds[wid][(lane & 15) * 32 + kpat(lane, j)];

        #pragma unroll
        for (int nt = 0; nt < 4; ++nt) {
            Frag vf;
            // V^T: column (dh) rows are contiguous in the key dimension
            const unsigned short* vcol =
                Vbh + (size_t)(nt * 16 + (lane & 15)) * 2048 + kt0;
            #pragma unroll
            for (int j = 0; j < 8; ++j)
                vf.u[j] = *(const unsigned int*)&vcol[kpat(lane, j)];
            o[nt] = wmma_bf16(pf, vf, o[nt]);
        }
    }

    // ---- normalize + store O as bf16 [B, T, C] ----
    const int b = bh >> 4, h = bh & 15;
    #pragma unroll
    for (int r = 0; r < 8; ++r) {
        float inv = 1.0f / lrow[r];
        int   t   = q0 + r + 8 * (lane >> 4);
        #pragma unroll
        for (int nt = 0; nt < 4; ++nt) {
            int c = h * 64 + nt * 16 + (lane & 15);
            O[((size_t)(b * 2048 + t)) * 1024 + c] = f2bf(o[nt][r] * inv);
        }
    }
}

// ---------------------------------------------------------------------------
extern "C" void kernel_launch(void* const* d_in, const int* in_sizes, int n_in,
                              void* d_out, int out_size, void* d_ws, size_t ws_size,
                              hipStream_t stream) {
    const float* x     = (const float*)d_in[0];   // [4, 2048, 1024]
    const float* w_qkv = (const float*)d_in[1];   // [1024, 3072]
    const float* w_out = (const float*)d_in[2];   // [1024, 1024]
    float*       out   = (float*)d_out;           // [4, 2048, 1024]

    const int B = 4, T = 2048, C = 1024;
    const int M = B * T;                          // 8192
    const size_t qkvElems = (size_t)B * 16 * T * 64;   // 8 Mi bf16 each

    unsigned short* qbuf  = (unsigned short*)d_ws;
    unsigned short* kbuf  = qbuf + qkvElems;
    unsigned short* vbufT = kbuf + qkvElems;      // transposed [B*H, 64, 2048]
    unsigned short* obuf  = vbufT + qkvElems;     // bf16 [B, T, C]

    // phase 1: qkv projection, scatter into Q/K/V^T (q pre-scaled)
    dim3 g1((3 * C) / 128, M / 128);              // 24 x 64
    gemm_bf16_wmma<0><<<g1, 256, 0, stream>>>(
        x, w_qkv, M, 3 * C, C, C, 3 * C, qbuf, kbuf, vbufT, nullptr);

    // phase 2: flash attention (8192 waves, 8 per block)
    attn_flash_wmma<<<1024, 256, 0, stream>>>(qbuf, kbuf, vbufT, obuf);

    // phase 3: output projection to fp32
    dim3 g3(C / 128, M / 128);                    // 8 x 64
    gemm_bf16_wmma<1><<<g3, 256, 0, stream>>>(
        obuf, w_out, M, C, C, C, C, nullptr, nullptr, nullptr, out);
}
